// Top1MoE_51161650430324
// MI455X (gfx1250) — matrix operand order repaired and checked by reading, not compiled, not run on hardware
//
#include <hip/hip_runtime.h>
#include <hip/hip_bf16.h>
#include <math.h>

#define D_MODEL    1024
#define EXPERT_DIM 4096
#define N_EXPERTS  8
#define NTOK       4096

typedef __attribute__((ext_vector_type(16))) __bf16 v16bf;
typedef __attribute__((ext_vector_type(8)))  __bf16 v8bf;
typedef __attribute__((ext_vector_type(8)))  float  v8f;
typedef __attribute__((ext_vector_type(4)))  float  v4f;

// Fast reciprocal (v_rcp_f32) -- avoids IEEE div_scale/div_fmas chains.
__device__ __forceinline__ float fast_rcp(float a) { return __builtin_amdgcn_rcpf(a); }

// Load 16 consecutive fp32 and convert to a packed v16bf (B-operand fragment).
__device__ __forceinline__ v16bf cvt16(const float* __restrict__ p) {
    v4f f0 = *(const v4f*)(p + 0);
    v4f f1 = *(const v4f*)(p + 4);
    v4f f2 = *(const v4f*)(p + 8);
    v4f f3 = *(const v4f*)(p + 12);
    v16bf r;
#pragma unroll
    for (int i = 0; i < 4; ++i) {
        r[i]      = (__bf16)f0[i];
        r[4 + i]  = (__bf16)f1[i];
        r[8 + i]  = (__bf16)f2[i];
        r[12 + i] = (__bf16)f3[i];
    }
    return r;
}

__global__ void moe_init_kernel(int* __restrict__ cnt, float* __restrict__ imp) {
    if (threadIdx.x < N_EXPERTS) { cnt[threadIdx.x] = 0; imp[threadIdx.x] = 0.f; }
}

// One wave32 per token: logits, softmax, top-1, routing lists, importance sums.
__global__ __launch_bounds__(256)
void moe_gate_kernel(const float* __restrict__ x, const float* __restrict__ gw,
                     float* __restrict__ top_p, int* __restrict__ cnt,
                     float* __restrict__ imp, int* __restrict__ list)
{
    const int wave = threadIdx.x >> 5;
    const int lane = threadIdx.x & 31;
    const int tok  = blockIdx.x * 8 + wave;
    if (tok >= NTOK) return;
    const float* xr = x + (size_t)tok * D_MODEL;

    float logit[N_EXPERTS];
#pragma unroll
    for (int e = 0; e < N_EXPERTS; ++e) {
        const float* g = gw + e * D_MODEL;
        float s = 0.f;
        for (int d = lane; d < D_MODEL; d += 32) s += xr[d] * g[d];
#pragma unroll
        for (int off = 16; off >= 1; off >>= 1) s += __shfl_xor(s, off, 32);
        logit[e] = s;   // full sum present in every lane after butterfly
    }
    float m = logit[0]; int bi = 0;
#pragma unroll
    for (int e = 1; e < N_EXPERTS; ++e)
        if (logit[e] > m) { m = logit[e]; bi = e; }   // strict > => first-index ties like argmax
    float pe[N_EXPERTS]; float se = 0.f;
#pragma unroll
    for (int e = 0; e < N_EXPERTS; ++e) { pe[e] = __expf(logit[e] - m); se += pe[e]; }
    const float inv = fast_rcp(se);
    if (lane == 0) {
        top_p[tok] = pe[bi] * inv;
#pragma unroll
        for (int e = 0; e < N_EXPERTS; ++e) atomicAdd(&imp[e], pe[e] * inv);
        int pos = atomicAdd(&cnt[bi], 1);
        list[bi * NTOK + pos] = tok;
    }
}

__global__ void moe_aux_kernel(const int* __restrict__ cnt, const float* __restrict__ imp,
                               float* __restrict__ out_aux) {
    float a = 0.f;
    for (int e = 0; e < N_EXPERTS; ++e) a += (float)cnt[e] * imp[e];
    *out_aux = a * ((float)N_EXPERTS * 0.01f) / ((float)NTOK * (float)NTOK);
}

// Expert FFN: per-expert 16-token tiles, bf16 WMMA, f32 accumulate.
__global__ __launch_bounds__(128)
void moe_ffn_kernel(const float* __restrict__ x,
                    const float* __restrict__ w1,
                    const float* __restrict__ w2,
                    const float* __restrict__ w3,
                    const int*   __restrict__ cnt,
                    const int*   __restrict__ list,
                    const float* __restrict__ top_p,
                    float*       __restrict__ out)
{
    __shared__ __align__(16) __bf16 Xs[16][D_MODEL + 8];   // +8 pad -> conflict-free b128 reads
    __shared__ __align__(16) __bf16 Hs[16][128 + 8];
    __shared__ float Ps[16];
    __shared__ int   Toks[16];

    const int e    = blockIdx.y;
    const int slot = blockIdx.x;
    const int tid  = threadIdx.x;
    const int wave = tid >> 5;
    const int lane = tid & 31;
    const int half = lane >> 4;     // 0: lanes 0-15, 1: lanes 16-31
    const int l15  = lane & 15;
    const int count = cnt[e];

    for (int tile0 = slot * 16; tile0 < count; tile0 += 32 * 16) {
        const int nt = min(16, count - tile0);
        if (tid < 16) {
            int tk = (tid < nt) ? list[e * NTOK + tile0 + tid] : 0;
            Toks[tid] = tk;
            Ps[tid]   = (tid < nt) ? top_p[tk] : 0.f;
        }
        __syncthreads();

        // Stage gathered X tile as bf16 (coalesced per row segment).
        for (int idx = tid; idx < 16 * D_MODEL; idx += 128) {
            int r = idx >> 10, c = idx & (D_MODEL - 1);
            float v = (r < nt) ? x[(size_t)Toks[r] * D_MODEL + c] : 0.f;
            Xs[r][c] = (__bf16)v;
        }
        __syncthreads();

        v8f acc[16];
#pragma unroll
        for (int t = 0; t < 16; ++t) acc[t] = (v8f){};

        for (int fb = 0; fb < EXPERT_DIM; fb += 128) {
            // ---- phase 1: this wave's 32 f-columns: u = X*w1^T, v = X*w3^T, h = silu(u)*v
#pragma unroll
            for (int sub = 0; sub < 2; ++sub) {
                const int fbase = fb + wave * 32 + sub * 16;
                v8f u = (v8f){}, vv = (v8f){};
                const float* w1p = w1 + ((size_t)e * EXPERT_DIM + fbase + l15) * D_MODEL;
                const float* w3p = w3 + ((size_t)e * EXPERT_DIM + fbase + l15) * D_MODEL;
                for (int k0 = 0; k0 < D_MODEL; k0 += 32) {
                    const int koff = k0 + 16 * half;
                    // Speculative prefetch ~2 iterations ahead of the streaming
                    // weight reads (lowers to global_prefetch_b8; OOB dropped).
                    __builtin_prefetch(w1p + koff + 64, 0, 0);
                    __builtin_prefetch(w3p + koff + 64, 0, 0);
                    // A fragment (contiguous k order, same as B): lanes hold row l15,
                    // K = k0 + 16*half + 0..15 (matches B)
                    v8bf alo = *(const v8bf*)&Xs[l15][k0 + 16 * half];
                    v8bf ahi = *(const v8bf*)&Xs[l15][k0 + 16 * half + 8];
                    v16bf A;
#pragma unroll
                    for (int i = 0; i < 8; ++i) { A[i] = alo[i]; A[8 + i] = ahi[i]; }
                    // B fragment (32x16): lane holds N=l15, K = 16*half + 0..15
                    v16bf B1 = cvt16(w1p + koff);
                    v16bf B3 = cvt16(w3p + koff);
                    u  = __builtin_amdgcn_wmma_f32_16x16x32_bf16(false, A, false, B1, (short)0, u,  false, false);
                    vv = __builtin_amdgcn_wmma_f32_16x16x32_bf16(false, A, false, B3, (short)0, vv, false, false);
                }
#pragma unroll
                for (int r = 0; r < 8; ++r) {
                    float uu = u[r];
                    // silu(u) = u * sigmoid(u), sigmoid via v_rcp_f32 (fast path)
                    float h  = uu * fast_rcp(1.f + __expf(-uu)) * vv[r];
                    int tok = r + 8 * half;   // C/D layout: M = r + 8*(lane>=16), N = l15
                    Hs[tok][wave * 32 + sub * 16 + l15] = (__bf16)h;
                }
            }
            __syncthreads();

            // ---- phase 2: out[:, wave's 256 dims] += h(16x128) * w2^T
            v16bf A2[4];
#pragma unroll
            for (int k = 0; k < 4; ++k) {
                v8bf alo = *(const v8bf*)&Hs[l15][k * 32 + 16 * half];
                v8bf ahi = *(const v8bf*)&Hs[l15][k * 32 + 16 * half + 8];
#pragma unroll
                for (int i = 0; i < 8; ++i) { A2[k][i] = alo[i]; A2[k][8 + i] = ahi[i]; }
            }
#pragma unroll
            for (int t = 0; t < 16; ++t) {
                const int d = (wave * 16 + t) * 16 + l15;
                const float* w2p = w2 + ((size_t)e * D_MODEL + d) * EXPERT_DIM + fb;
                __builtin_prefetch(w2p + 128, 0, 0);   // next fb block of this row
#pragma unroll
                for (int k = 0; k < 4; ++k) {
                    v16bf B2 = cvt16(w2p + k * 32 + 16 * half);
                    acc[t] = __builtin_amdgcn_wmma_f32_16x16x32_bf16(false, A2[k], false, B2, (short)0, acc[t], false, false);
                }
            }
            __syncthreads();
        }

        // Epilogue: scale by gate prob, scatter rows.
#pragma unroll
        for (int t = 0; t < 16; ++t) {
            const int d = (wave * 16 + t) * 16 + l15;
#pragma unroll
            for (int r = 0; r < 8; ++r) {
                int tok = r + 8 * half;
                if (tok < nt)
                    out[(size_t)Toks[tok] * D_MODEL + d] = acc[t][r] * Ps[tok];
            }
        }
        __syncthreads();
    }
}

extern "C" void kernel_launch(void* const* d_in, const int* in_sizes, int n_in,
                              void* d_out, int out_size, void* d_ws, size_t ws_size,
                              hipStream_t stream) {
    const float* x  = (const float*)d_in[0];
    const float* gw = (const float*)d_in[1];
    const float* w1 = (const float*)d_in[2];
    const float* w2 = (const float*)d_in[3];
    const float* w3 = (const float*)d_in[4];
    float* out = (float*)d_out;

    char*  ws    = (char*)d_ws;
    float* top_p = (float*)ws;                         // 4096 f32
    int*   cnt   = (int*)  (ws + NTOK * 4);            // 8 int
    float* imp   = (float*)(ws + NTOK * 4 + 32);       // 8 f32
    int*   list  = (int*)  (ws + NTOK * 4 + 64);       // 8 * 4096 int

    moe_init_kernel<<<1, 32, 0, stream>>>(cnt, imp);
    moe_gate_kernel<<<NTOK / 8, 256, 0, stream>>>(x, gw, top_p, cnt, imp, list);
    moe_aux_kernel<<<1, 1, 0, stream>>>(cnt, imp, out + (size_t)NTOK * D_MODEL);
    moe_ffn_kernel<<<dim3(32, N_EXPERTS), 128, 0, stream>>>(x, w1, w2, w3, cnt, list, top_p, out);
}
